// SimilarityLoss_72937134620697
// MI455X (gfx1250) — matrix operand order repaired and checked by reading, not compiled, not run on hardware
//
#include <hip/hip_runtime.h>
#include <hip/hip_bf16.h>
#include <math.h>

typedef __attribute__((ext_vector_type(16))) __bf16 v16bf;
typedef __attribute__((ext_vector_type(8)))  float  v8f;

#define B_   8
#define M_   36
#define D_   512
#define HW_  196
#define N_   7056
#define NP_  7072        // N padded to multiple of 32 for WMMA K
#define T_   64
#define NPAIR 64
#define G1 4.0f
#define G2 5.0f
#define G3 10.0f

// ---- workspace layout (bytes) ----
#define OFF_V    ((size_t)0)
#define SZ_V     ((size_t)B_*NP_*D_*2)          // bf16 v (permuted image), padded rows
#define OFF_E    (OFF_V + SZ_V)
#define SZ_E     ((size_t)B_*T_*D_*2)           // bf16 text
#define OFF_W    (OFF_E + SZ_E)
#define SZ_W     ((size_t)NPAIR*T_*NP_*2)       // bf16 exp(4*norm_sim) per pair
#define OFF_Z1   (OFF_W + SZ_W)
#define SZ_Z1    ((size_t)NPAIR*T_*4)           // attention denominators
#define OFF_BS   (OFF_Z1 + SZ_Z1)
#define SZ_BS    ((size_t)NPAIR*T_*M_*4)        // per-box exp(norm_sim) sums
#define ZERO_END (OFF_BS + SZ_BS)
#define OFF_VT   ZERO_END
#define SZ_VT    ((size_t)NPAIR*T_*D_*4)        // v_tidal_raw (unnormalized)
#define OFF_EP   (OFF_VT + SZ_VT)
#define SZ_EP    ((size_t)NPAIR*D_*M_*4)        // e_prime
#define OFF_RQD  (OFF_EP + SZ_EP)
#define SZ_RQD   ((size_t)256)
#define OFF_REF  (OFF_RQD + SZ_RQD)
#define SZ_REF   ((size_t)NPAIR*M_*4)

// ---------------- WMMA fragment helpers (wave32, contiguous k order used in both A and B) ----------------
// A-matrix 16x32 (MxK): lanes 0-15 M=0..15 kb=0, lanes 16-31 M=0..15 kb=16.
// elements 0..15 -> K=kb+0..15
__device__ inline v16bf load_a_frag(const __bf16* A, int ld, int row0, int k0) {
  int lane = threadIdx.x & 31;
  int m    = lane & 15;
  int kb   = (lane < 16) ? 0 : 16;
  const __bf16* p = A + (size_t)(row0 + m) * ld + (k0 + kb);
  v16bf f;
#pragma unroll
  for (int j = 0; j < 16; ++j) f[j] = p[j];
  return f;
}

// B-matrix 32x16 (KxN) where source stores B[k][n] = V[n][k] (row n contiguous in k):
// lane n = lane&15, kb = (lane<16)?0:16, element j -> K = kb + j  (contiguous 32B load)
__device__ inline v16bf load_b_frag_rows(const __bf16* V, int ldv, int n0, int k0) {
  int lane = threadIdx.x & 31;
  int n    = n0 + (lane & 15);
  int kb   = (lane < 16) ? 0 : 16;
  const __bf16* p = V + (size_t)n * ldv + (k0 + kb);
  v16bf f;
#pragma unroll
  for (int j = 0; j < 16; ++j) f[j] = p[j];
  return f;
}

__device__ inline v8f wmma_bf16(v16bf a, v16bf b, v8f c) {
  return __builtin_amdgcn_wmma_f32_16x16x32_bf16(false, a, false, b, (short)0, c,
                                                 false, false);
}

// async global->LDS copy of 16 bytes (GLOBAL_LOAD_ASYNC_TO_LDS_B128, ASYNCcnt path)
__device__ inline void async_copy_b128(void* lds_dst, const void* gsrc) {
  unsigned lds = (unsigned)(size_t)lds_dst;  // LDS offset = addr[31:0]
  unsigned long long ga = (unsigned long long)(size_t)gsrc;
  asm volatile("global_load_async_to_lds_b128 %0, %1, off"
               :: "v"(lds), "v"(ga)
               : "memory");
}

__device__ inline void wait_asynccnt0() {
  asm volatile("s_wait_asynccnt 0x0" ::: "memory");
}

// ---------------- utility kernels ----------------
__global__ void k_zero(uint4* p, size_t n16) {
  size_t i = (size_t)blockIdx.x * blockDim.x + threadIdx.x;
  size_t st = (size_t)gridDim.x * blockDim.x;
  uint4 z; z.x = z.y = z.z = z.w = 0u;
  for (; i < n16; i += st) p[i] = z;
}

// v[b, m*196 + w*14 + h, d] = image[b, m, d, h, w]   (permute (0,1,4,3,2))
__global__ void k_build_v(const float* __restrict__ img, __bf16* __restrict__ v) {
  size_t idx = (size_t)blockIdx.x * blockDim.x + threadIdx.x;
  if (idx >= (size_t)B_ * N_ * D_) return;
  int d = (int)(idx % D_);
  size_t r = idx / D_;
  int n = (int)(r % N_);
  int b = (int)(r / N_);
  int m = n / HW_;
  int rem = n % HW_;
  int w = rem / 14;
  int h = rem % 14;
  size_t src = ((((size_t)b * M_ + m) * D_ + d) * 14 + h) * 14 + w;
  v[((size_t)b * NP_ + n) * D_ + d] = (__bf16)img[src];
}

__global__ void k_build_e(const float* __restrict__ t, __bf16* __restrict__ e) {
  size_t idx = (size_t)blockIdx.x * blockDim.x + threadIdx.x;
  if (idx >= (size_t)B_ * T_ * D_) return;
  e[idx] = (__bf16)t[idx];
}

// ---------------- kernel 1: sim tiles + word-softmax + W/Z1/bsum ----------------
// grid (63, 64 pairs), block 128 (4 waves). Each WG: 64x112 sim block via WMMA.
__global__ __launch_bounds__(128) void k_sim(const __bf16* __restrict__ vmat,
                                             const __bf16* __restrict__ emat,
                                             __bf16* __restrict__ W,
                                             float* __restrict__ Z1,
                                             float* __restrict__ bsum) {
  __shared__ float ls[T_][113];
  __shared__ float cmax[112];
  __shared__ float csum[112];
  int pair = blockIdx.y;
  int i = pair >> 3;          // image index
  int j = pair & 7;           // text index
  int cb = blockIdx.x * 112;  // column base in N
  int wave = threadIdx.x >> 5;
  int lane = threadIdx.x & 31;
  int tid = threadIdx.x;
  int t0 = wave * 16;

  const __bf16* Ve = vmat + (size_t)i * NP_ * D_;
  const __bf16* Ee = emat + (size_t)j * T_ * D_;

  v8f acc[7];
  v8f zf = {0.f, 0.f, 0.f, 0.f, 0.f, 0.f, 0.f, 0.f};
#pragma unroll
  for (int c = 0; c < 7; ++c) acc[c] = zf;

  // sim[t][n] = sum_d e[t][d] * v[n][d]
  for (int kc = 0; kc < D_ / 32; ++kc) {
    // prefetch next k-chunk of the 112 B-rows (one row per thread, uniform branch)
    if (kc + 1 < D_ / 32) {
      int pr = tid < 112 ? tid : 0;
      __builtin_prefetch(Ve + (size_t)(cb + pr) * D_ + (kc + 1) * 32, 0, 1);
    }
    v16bf a = load_a_frag(Ee, D_, t0, kc * 32);
#pragma unroll
    for (int c = 0; c < 7; ++c) {
      v16bf b = load_b_frag_rows(Ve, D_, cb + c * 16, kc * 32);
      acc[c] = wmma_bf16(a, b, acc[c]);
    }
  }

  int lhalf = (lane < 16) ? 0 : 8;
  int lcol = lane & 15;
#pragma unroll
  for (int c = 0; c < 7; ++c)
#pragma unroll
    for (int r = 0; r < 8; ++r)
      ls[t0 + lhalf + r][c * 16 + lcol] = acc[c][r];
  __syncthreads();

  // column softmax over words (axis=0, T=64 rows all resident)
  if (tid < 112) {
    float mx = -1e30f;
    for (int t = 0; t < T_; ++t) mx = fmaxf(mx, ls[t][tid]);
    float s = 0.f;
    for (int t = 0; t < T_; ++t) s += __expf(ls[t][tid] - mx);
    cmax[tid] = mx;
    csum[tid] = s;
  }
  __syncthreads();

  // per-row accumulations: Z1 = sum exp(4*ns), bsum per box = sum exp(ns)
  if (tid < T_) {
    int m0 = cb / HW_;
    int start2 = (m0 + 1) * HW_ - cb;  // col where next box starts (tile spans <=2 boxes)
    float z1 = 0.f, b0 = 0.f, b1 = 0.f;
    for (int c = 0; c < 112; ++c) {
      float ns = __expf(ls[tid][c] - cmax[c]) / csum[c];  // in (0,1): exp safe
      z1 += __expf(G1 * ns);
      float en = __expf(ns);
      if (c < start2) b0 += en; else b1 += en;
    }
    atomicAdd(&Z1[pair * T_ + tid], z1);
    atomicAdd(&bsum[(pair * T_ + tid) * M_ + m0], b0);
    if (start2 < 112) atomicAdd(&bsum[(pair * T_ + tid) * M_ + m0 + 1], b1);
  }

  // write W = bf16(exp(4*ns)) coalesced, A-matrix for second matmul
  __bf16* Wp = W + (size_t)pair * T_ * NP_;
  for (int idx = tid; idx < T_ * 112; idx += 128) {
    int t = idx / 112;
    int c = idx % 112;
    float ns = __expf(ls[t][c] - cmax[c]) / csum[c];
    Wp[(size_t)t * NP_ + cb + c] = (__bf16)__expf(G1 * ns);
  }
}

// ---------------- kernel 2: v_tidal_raw = W @ v  (64x512, K=7072) ----------------
// grid (4, 64 pairs), block 128 (4 waves); v k-chunk DMA'd into LDS via
// GLOBAL_LOAD_ASYNC_TO_LDS_B128 (ASYNCcnt), bypassing VGPRs.
__global__ __launch_bounds__(128) void k_vtidal(const __bf16* __restrict__ vmat,
                                                const __bf16* __restrict__ W,
                                                float* __restrict__ vtr) {
  __shared__ __align__(16) __bf16 lv[32][136];  // row stride 272B = 17*16: 16B aligned
  int pair = blockIdx.y;
  int i = pair >> 3;
  int dblock = blockIdx.x * 128;
  int wave = threadIdx.x >> 5;
  int lane = threadIdx.x & 31;
  int tid = threadIdx.x;
  int t0 = wave * 16;

  const __bf16* Ve = vmat + (size_t)i * NP_ * D_;
  const __bf16* Wp = W + (size_t)pair * T_ * NP_;

  v8f acc[8];
  v8f zf = {0.f, 0.f, 0.f, 0.f, 0.f, 0.f, 0.f, 0.f};
#pragma unroll
  for (int c = 0; c < 8; ++c) acc[c] = zf;

  int kb = (lane < 16) ? 0 : 16;
  int dl = lane & 15;

  for (int kc = 0; kc < NP_ / 32; ++kc) {
    int k0 = kc * 32;
    __syncthreads();
    // async DMA: v[k0..k0+31][dblock..dblock+127] -> LDS, 16B per issue, 4 per thread
#pragma unroll
    for (int s = 0; s < 4; ++s) {
      int seg = tid + s * 128;        // 512 segments of 16B
      int r = seg >> 4;               // row 0..31
      int ce = (seg & 15) * 8;        // col element (8 bf16)
      async_copy_b128(&lv[r][ce], Ve + (size_t)(k0 + r) * D_ + dblock + ce);
    }
    // prefetch next chunk into cache while DMA completes (uniform branch)
    if (k0 + 32 < NP_) {
      __builtin_prefetch(Ve + (size_t)(k0 + 32 + (tid & 31)) * D_ + dblock, 0, 1);
    }
    wait_asynccnt0();
    __syncthreads();
    v16bf a = load_a_frag(Wp, NP_, t0, k0);
#pragma unroll
    for (int ct = 0; ct < 8; ++ct) {
      v16bf b;
#pragma unroll
      for (int jj = 0; jj < 16; ++jj) b[jj] = lv[kb + jj][ct * 16 + dl];
      acc[ct] = wmma_bf16(a, b, acc[ct]);
    }
  }

  int lhalf = (lane < 16) ? 0 : 8;
  float* out = vtr + (size_t)pair * T_ * D_;
#pragma unroll
  for (int ct = 0; ct < 8; ++ct)
#pragma unroll
    for (int r = 0; r < 8; ++r)
      out[(size_t)(t0 + lhalf + r) * D_ + dblock + ct * 16 + dl] = acc[ct][r];
}

// ---------------- kernel 3a: beta + e_prime ----------------
__global__ __launch_bounds__(256) void k_beta_eprime(const float* __restrict__ text,
                                                     const float* __restrict__ bsum,
                                                     float* __restrict__ eprime) {
  __shared__ float lsum[M_];
  __shared__ float lbeta[T_][M_];
  int pair = blockIdx.x;
  int j = pair & 7;
  int tid = threadIdx.x;
  const float* bs = bsum + (size_t)pair * T_ * M_;
  if (tid < M_) {
    float s = 0.f;
    for (int t = 0; t < T_; ++t) s += bs[t * M_ + tid];
    lsum[tid] = s;
  }
  __syncthreads();
  for (int idx = tid; idx < T_ * M_; idx += 256) {
    int t = idx / M_, m = idx % M_;
    lbeta[t][m] = bs[t * M_ + m] / lsum[m];
  }
  __syncthreads();
  const float* ej = text + (size_t)j * T_ * D_;
  float* ep = eprime + (size_t)pair * D_ * M_;
  for (int idx = tid; idx < D_ * M_; idx += 256) {
    int d = idx / M_, m = idx % M_;
    float s = 0.f;
    for (int t = 0; t < T_; ++t) s += ej[(size_t)t * D_ + d] * lbeta[t][m];
    ep[idx] = s;
  }
}

// ---------------- kernel 3b: r_qd per pair ----------------
__global__ __launch_bounds__(64) void k_rqd(const float* __restrict__ text,
                                            const float* __restrict__ vtr,
                                            const float* __restrict__ Z1,
                                            float* __restrict__ rqd) {
  __shared__ float red[T_];
  int pair = blockIdx.x;
  int j = pair & 7;
  int t = threadIdx.x;
  const float* ej = text + ((size_t)j * T_ + t) * D_;
  const float* vt = vtr + ((size_t)pair * T_ + t) * D_;
  float s = 0.f;
  for (int d = 0; d < D_; ++d) s += vt[d] * ej[d];
  float dg = s / Z1[pair * T_ + t];
  red[t] = __expf(G2 * dg);
  __syncthreads();
  if (t == 0) {
    float a = 0.f;
    for (int k = 0; k < T_; ++k) a += red[k];
    rqd[pair] = (1.f / G2) * __logf(a);
  }
}

// ---------------- kernel 3c: cosine refs per (pair, box) from fp32 image ----------------
__global__ __launch_bounds__(256) void k_ref(const float* __restrict__ img,
                                             const float* __restrict__ eprime,
                                             float* __restrict__ refs) {
  __shared__ float red[256];
  __shared__ float epl[D_];
  int m = blockIdx.x;
  int pair = blockIdx.y;
  int i = pair >> 3;
  int tid = threadIdx.x;
  const float* ep = eprime + (size_t)pair * D_ * M_;
  for (int d = tid; d < D_; d += 256) epl[d] = ep[(size_t)d * M_ + m];
  __syncthreads();
  float p = 0.f;
  for (int d = tid; d < D_; d += 256) p += epl[d] * epl[d];
  red[tid] = p;
  __syncthreads();
  for (int s = 128; s > 0; s >>= 1) {
    if (tid < s) red[tid] += red[tid + s];
    __syncthreads();
  }
  float epn = sqrtf(red[0]);
  __syncthreads();
  float val = 0.f;
  if (tid < HW_) {
    int w = tid / 14, h = tid % 14;  // n = m*196 + w*14 + h
    const float* base = img + (size_t)(i * M_ + m) * D_ * 196 + h * 14 + w;
    float dot = 0.f, nrm = 0.f;
    for (int d = 0; d < D_; ++d) {
      float x = base[(size_t)d * 196];
      dot += x * epl[d];
      nrm += x * x;
    }
    val = dot / (sqrtf(nrm) * epn);
  }
  red[tid] = val;
  __syncthreads();
  for (int s = 128; s > 0; s >>= 1) {
    if (tid < s) red[tid] += red[tid + s];
    __syncthreads();
  }
  if (tid == 0) refs[pair * M_ + m] = red[0] / (float)HW_;
}

// ---------------- kernel 4: final losses ----------------
__global__ __launch_bounds__(256) void k_final(const float* __restrict__ rqd,
                                               const float* __restrict__ refs,
                                               const float* __restrict__ bsum,
                                               const int* __restrict__ spinds,
                                               float* __restrict__ out) {
  __shared__ float S[NPAIR];
  __shared__ float lsum[M_];
  __shared__ float red1[256];
  __shared__ float red2[256];
  __shared__ float losses;
  __shared__ float regacc;
  int tid = threadIdx.x;
  if (tid < NPAIR) {
    float a = 0.f;
    for (int m = 0; m < M_; ++m) a += __expf(refs[tid * M_ + m]);
    S[tid] = rqd[tid] + (1.f / G2) * __logf(a);
  }
  __syncthreads();
  if (tid == 0) {
    float l1 = 0.f, l2 = 0.f;
    for (int b = 0; b < B_; ++b) {
      float num = G3 * __expf(S[b * B_ + b]);
      float pdq = 0.f, pqd = 0.f;
      for (int k = 0; k < 5; ++k) {
        int sp = spinds[b * 5 + k];
        pdq += G3 * __expf(S[b * B_ + sp]);
        pqd += G3 * __expf(S[sp * B_ + b]);
      }
      l1 += num / pdq;
      l2 += num / pqd;
    }
    losses = -(l1 / B_) - (l2 / B_);
    regacc = 0.f;
  }
  for (int b = 0; b < B_; ++b) {
    const float* bs = bsum + (size_t)(b * B_ + b) * T_ * M_;
    __syncthreads();
    if (tid < M_) {
      float s = 0.f;
      for (int t = 0; t < T_; ++t) s += bs[t * M_ + tid];
      lsum[tid] = s;
    }
    __syncthreads();
    float ss = 0.f, dd = 0.f;
    for (int idx = tid; idx < T_ * T_; idx += 256) {
      int t = idx / T_, s2 = idx % T_;
      float g = 0.f;
      for (int m = 0; m < M_; ++m)
        g += (bs[t * M_ + m] / lsum[m]) * (bs[s2 * M_ + m] / lsum[m]);
      ss += g * g;
      if (t == s2) dd += g * g;
    }
    red1[tid] = ss;
    red2[tid] = dd;
    __syncthreads();
    for (int s = 128; s > 0; s >>= 1) {
      if (tid < s) { red1[tid] += red1[tid + s]; red2[tid] += red2[tid + s]; }
      __syncthreads();
    }
    if (tid == 0) regacc += sqrtf(fmaxf(red1[0] - red2[0], 0.f));
    __syncthreads();
  }
  if (tid == 0) out[0] = losses + regacc / (float)(B_ * T_ * M_);
}

extern "C" void kernel_launch(void* const* d_in, const int* in_sizes, int n_in,
                              void* d_out, int out_size, void* d_ws, size_t ws_size,
                              hipStream_t stream) {
  const float* img = (const float*)d_in[0];
  const float* text = (const float*)d_in[1];
  const int* spinds = (const int*)d_in[3];
  float* out = (float*)d_out;

  char* ws = (char*)d_ws;
  __bf16* vb = (__bf16*)(ws + OFF_V);
  __bf16* eb = (__bf16*)(ws + OFF_E);
  __bf16* W = (__bf16*)(ws + OFF_W);
  float* Z1 = (float*)(ws + OFF_Z1);
  float* bsum = (float*)(ws + OFF_BS);
  float* vtr = (float*)(ws + OFF_VT);
  float* eprime = (float*)(ws + OFF_EP);
  float* rqd = (float*)(ws + OFF_RQD);
  float* refs = (float*)(ws + OFF_REF);

  // 0) zero accumulators + padded regions (v pad rows, W pad cols, Z1, bsum)
  size_t n16 = ZERO_END / 16;
  k_zero<<<4096, 256, 0, stream>>>((uint4*)ws, n16);

  // 1) precision conversion / permutation
  size_t nv = (size_t)B_ * N_ * D_;
  k_build_v<<<(unsigned)((nv + 255) / 256), 256, 0, stream>>>(img, vb);
  size_t ne = (size_t)B_ * T_ * D_;
  k_build_e<<<(unsigned)((ne + 255) / 256), 256, 0, stream>>>(text, eb);

  // 2) sim + word-softmax + W/Z1/bsum (heavy WMMA)
  k_sim<<<dim3(N_ / 112, NPAIR), 128, 0, stream>>>(vb, eb, W, Z1, bsum);

  // 3) v_tidal_raw = W @ v (heavy WMMA, async-LDS staging)
  k_vtidal<<<dim3(D_ / 128, NPAIR), 128, 0, stream>>>(vb, W, vtr);

  // 4) small stages
  k_beta_eprime<<<NPAIR, 256, 0, stream>>>(text, bsum, eprime);
  k_rqd<<<NPAIR, 64, 0, stream>>>(text, vtr, Z1, rqd);
  k_ref<<<dim3(M_, NPAIR), 256, 0, stream>>>(img, eprime, refs);

  // 5) finalize losses
  k_final<<<1, 256, 0, stream>>>(rqd, refs, bsum, spinds, out);

  (void)in_sizes; (void)n_in; (void)out_size; (void)ws_size;
}
